// SDPGNP_10247791968297
// MI455X (gfx1250) — matrix operand order repaired and checked by reading, not compiled, not run on hardware
//
#include <hip/hip_runtime.h>
#include <hip/hip_bf16.h>
#include <math.h>

// ---------------- problem constants (match reference) ----------------
#define NBS     32
#define N_NODE  400
#define NN      12800          // N = BS*N_NODE
#define EE      204800         // E
#define ETOT    217600         // E + N (self loops)
#define DD      200
#define HSZ     1024
#define SEQL    512
#define NTYPES  4
#define ETYPES  38
#define NHEADS  4
#define DPH     50
#define KLAYERS 5

#define NP      224            // N (=200) padded to multiple of BN
#define KP200   224            // K=200 padded to multiple of BK
#define KP400   416            // K=400 padded to multiple of BK

typedef __bf16 bf16_t;
typedef bf16_t v16bf __attribute__((ext_vector_type(16)));
typedef float  v8f   __attribute__((ext_vector_type(8)));

// ---------------- WMMA bf16 GEMM: C = act(A @ B + bias) ----------------
// A:  [M,K] bf16 row-major (lda);  REQUIRES M % BM == 0 (holds: 217600, 12800)
// Bt: [NP][KP] bf16, B transposed + zero-padded (cols>=N and k>=K are 0)
// Cf (fp32) and/or Cbf (bf16) outputs, ldc = N pitch. act: 0=none, 1=relu
#define BM 256
#define BN 32
#define BK 32

__device__ inline void store_tile(v8f acc, int cm0, int cn, int N,
                                  const float* __restrict__ bias,
                                  float* __restrict__ Cf, bf16_t* __restrict__ Cbf,
                                  int ldc, int act)
{
    if (cn >= N) return;
    float bv = bias ? bias[cn] : 0.0f;
    #pragma unroll
    for (int r = 0; r < 8; ++r) {
        int cm = cm0 + r;
        float v = acc[r] + bv;
        if (act == 1) v = fmaxf(v, 0.0f);
        if (Cf)  Cf [(size_t)cm * ldc + cn] = v;
        if (Cbf) Cbf[(size_t)cm * ldc + cn] = (bf16_t)v;
    }
}

template<int K>
__global__ __launch_bounds__(256)
void gemm_bf16_wmma(const bf16_t* __restrict__ A, int lda,
                    const bf16_t* __restrict__ Bt,
                    const float* __restrict__ bias,
                    float* __restrict__ Cf, bf16_t* __restrict__ Cbf, int ldc,
                    int N, int act)
{
    constexpr int NSTEPS = (K + BK - 1) / BK;
    constexpr int KPAD   = NSTEPS * BK;

    __shared__ __align__(32) bf16_t As[2][BM][BK];   // 2 x 16 KB double buffer

    const int t    = threadIdx.x;
    const int lane = t & 31;
    const int wave = t >> 5;          // 0..7 -> 32-row m-slab
    const int bM   = blockIdx.x * BM;
    const int bN   = blockIdx.y * BN;

    const bf16_t* Arow = A + (size_t)(bM + t) * lda;   // one 32-elem row per thread

    const int r  = lane & 15;
    const int hi = lane >> 4;

    // direct-from-global B fragment base pointers (contiguous 32B per lane,
    // in-bounds by construction thanks to NP/KP zero padding)
    const bf16_t* bt0 = Bt + (size_t)(bN + r)      * KPAD + hi * 16;
    const bf16_t* bt1 = Bt + (size_t)(bN + 16 + r) * KPAD + hi * 16;

    v8f acc00 = {}, acc01 = {}, acc10 = {}, acc11 = {};

    // stage A tile `step` into LDS buffer `buf`; all guards are compile-time
    auto stageA = [&](int step, int buf) {
        const int k0 = step * BK;
        uint4 av[4];
        #pragma unroll
        for (int c = 0; c < 4; ++c)
            av[c] = (k0 + c * 8 + 8 <= K)
                  ? *reinterpret_cast<const uint4*>(Arow + k0 + c * 8)
                  : (uint4){0u, 0u, 0u, 0u};
        if (k0 + 2 * BK < K)
            __builtin_prefetch(Arow + k0 + 2 * BK, 0, 3);   // tile after next
        #pragma unroll
        for (int c = 0; c < 4; ++c)
            *reinterpret_cast<uint4*>(&As[buf][t][c * 8]) = av[c];
    };

    stageA(0, 0);
    __syncthreads();

    #pragma unroll
    for (int i = 0; i < NSTEPS; ++i) {
        const int cur = i & 1;
        const int k0  = i * BK;
        // stage next A tile into the other buffer (nobody reads it this iter)
        if (i + 1 < NSTEPS) stageA(i + 1, cur ^ 1);

        // ---- A fragments from LDS (contiguous k order, same as B fragments) ----
        // 16x32: lanes0-15 row=l, K{0..15}; lanes16-31 row=l-16, K{16..31}
        v16bf a0, a1, b0, b1;
        {
            const uint4* ap0 = reinterpret_cast<const uint4*>(&As[cur][wave * 32 + r][0]);
            reinterpret_cast<uint4*>(&a0)[0] = ap0[hi ? 2 : 0];
            reinterpret_cast<uint4*>(&a0)[1] = ap0[hi ? 3 : 1];
            const uint4* ap1 = reinterpret_cast<const uint4*>(&As[cur][wave * 32 + 16 + r][0]);
            reinterpret_cast<uint4*>(&a1)[0] = ap1[hi ? 2 : 0];
            reinterpret_cast<uint4*>(&a1)[1] = ap1[hi ? 3 : 1];
        }
        // ---- B fragments straight from global (L2-resident weights) ----
        // 32x16: lanes0-15 col=l hold K k0..k0+15; lanes16-31 col=l-16 hold +16
        {
            reinterpret_cast<uint4*>(&b0)[0] = *reinterpret_cast<const uint4*>(bt0 + k0);
            reinterpret_cast<uint4*>(&b0)[1] = *reinterpret_cast<const uint4*>(bt0 + k0 + 8);
            reinterpret_cast<uint4*>(&b1)[0] = *reinterpret_cast<const uint4*>(bt1 + k0);
            reinterpret_cast<uint4*>(&b1)[1] = *reinterpret_cast<const uint4*>(bt1 + k0 + 8);
        }
        // 2x2 accumulator grid: 4 WMMAs reuse 2 A-frags and 2 B-frags
        acc00 = __builtin_amdgcn_wmma_f32_16x16x32_bf16(false, a0, false, b0, (short)0, acc00, false, false);
        acc01 = __builtin_amdgcn_wmma_f32_16x16x32_bf16(false, a0, false, b1, (short)0, acc01, false, false);
        acc10 = __builtin_amdgcn_wmma_f32_16x16x32_bf16(false, a1, false, b0, (short)0, acc10, false, false);
        acc11 = __builtin_amdgcn_wmma_f32_16x16x32_bf16(false, a1, false, b1, (short)0, acc11, false, false);
        __syncthreads();
    }

    // ---- epilogue: C 16x16 layout: VGPR r -> M = r (+8 for lanes>=16), N = lane&15 ----
    const int cn0 = bN + (lane & 15);
    const int rb  = bM + wave * 32 + (hi ? 8 : 0);
    store_tile(acc00, rb,      cn0,      N, bias, Cf, Cbf, ldc, act);
    store_tile(acc01, rb,      cn0 + 16, N, bias, Cf, Cbf, ldc, act);
    store_tile(acc10, rb + 16, cn0,      N, bias, Cf, Cbf, ldc, act);
    store_tile(acc11, rb + 16, cn0 + 16, N, bias, Cf, Cbf, ldc, act);
}

// ---------------- helpers ----------------
__device__ inline unsigned enc_f32(float f) {
    unsigned u = __float_as_uint(f);
    return (u & 0x80000000u) ? ~u : (u | 0x80000000u);
}
__device__ inline float dec_f32(unsigned e) {
    unsigned u = (e & 0x80000000u) ? (e & 0x7fffffffu) : ~e;
    return __uint_as_float(u);
}
__device__ inline unsigned hash_u32(unsigned x) {
    x ^= x >> 16; x *= 0x7feb352dU; x ^= x >> 15; x *= 0x846ca68bU; x ^= x >> 16;
    return x;
}
__device__ inline int edge_src(const int* ei, int e) { return (e < EE) ? ei[e] : (e - EE); }
__device__ inline int edge_dst(const int* ei, int e) { return (e < EE) ? ei[EE + e] : (e - EE); }

// ---------------- elementwise / glue kernels ----------------
__global__ void k_fill_f32(float* p, float v, size_t n) {
    size_t i = (size_t)blockIdx.x * blockDim.x + threadIdx.x;
    if (i < n) p[i] = v;
}
__global__ void k_fill_u32(unsigned* p, unsigned v, size_t n) {
    size_t i = (size_t)blockIdx.x * blockDim.x + threadIdx.x;
    if (i < n) p[i] = v;
}
__global__ void k_copy_f32(const float* __restrict__ in, float* __restrict__ out, size_t n) {
    size_t i = (size_t)blockIdx.x * blockDim.x + threadIdx.x;
    if (i < n) out[i] = in[i];
}
__global__ void k_f32_to_bf16(const float* __restrict__ in, bf16_t* __restrict__ out, size_t n) {
    size_t i = (size_t)blockIdx.x * blockDim.x + threadIdx.x;
    if (i < n) out[i] = (bf16_t)in[i];
}

// Bt[n][k] = (n<N && k<K) ? bf16(W[k*N+n]) : 0 over NP x KP (transpose + pad)
__global__ void k_transpose_pad(const float* __restrict__ W, bf16_t* __restrict__ Bt,
                                int K, int N, int KPAD, size_t total)
{
    size_t i = (size_t)blockIdx.x * blockDim.x + threadIdx.x;
    if (i >= total) return;
    int n = (int)(i / KPAD), k = (int)(i % KPAD);
    float v = (n < N && k < K) ? W[(size_t)k * N + n] : 0.0f;
    Bt[i] = (bf16_t)v;
}

// Edge encoder layer 1 is a one-hot matmul == 3-row gather of We1 + bias, relu, bf16
__global__ void k_edge_encode(const int* __restrict__ ei, const int* __restrict__ etype,
                              const int* __restrict__ ntype,
                              const float* __restrict__ We1, const float* __restrict__ be1,
                              bf16_t* __restrict__ H1)
{
    size_t i = (size_t)blockIdx.x * blockDim.x + threadIdx.x;
    if (i >= (size_t)ETOT * DD) return;
    int e = (int)(i / DD), d = (int)(i % DD);
    int s, t2, et;
    if (e < EE) { s = ei[e]; t2 = ei[EE + e]; et = etype[e]; }
    else        { s = t2 = e - EE; et = ETYPES; }
    int ht = ntype[s], tt = ntype[t2];
    float v = We1[(size_t)et * DD + d]
            + We1[(size_t)(ETYPES + 1 + ht) * DD + d]
            + We1[(size_t)(ETYPES + 1 + NTYPES + tt) * DD + d]
            + be1[d];
    H1[i] = (bf16_t)fmaxf(v, 0.0f);
}

// xcat_bf16[n, 0:200] = X, [200:400] = node_feature_extra
__global__ void k_xcat(const float* __restrict__ X, const float* __restrict__ nfe,
                       bf16_t* __restrict__ xcat)
{
    size_t i = (size_t)blockIdx.x * blockDim.x + threadIdx.x;
    if (i >= (size_t)NN * 2 * DD) return;
    int n = (int)(i / (2 * DD)), d = (int)(i % (2 * DD));
    float v = (d < DD) ? X[(size_t)n * DD + d] : nfe[(size_t)n * DD + (d - DD)];
    xcat[i] = (bf16_t)v;
}

// scores[e,h] = (qryn[dst] . (keyn[dst] + keye[e]))_h / sqrt(DPH)
__global__ void k_score(const int* __restrict__ ei, const float* __restrict__ qryn,
                        const float* __restrict__ keyn, const bf16_t* __restrict__ keye,
                        float* __restrict__ score)
{
    size_t i = (size_t)blockIdx.x * blockDim.x + threadIdx.x;
    if (i >= (size_t)ETOT * NHEADS) return;
    int e = (int)(i >> 2), h = (int)(i & 3);
    int dst = edge_dst(ei, e);
    const float*  q  = qryn + (size_t)dst * DD + h * DPH;
    const float*  kn = keyn + (size_t)dst * DD + h * DPH;
    const bf16_t* ke = keye + (size_t)e   * DD + h * DPH;
    float s = 0.0f;
    #pragma unroll 10
    for (int d = 0; d < DPH; ++d) s += q[d] * (kn[d] + (float)ke[d]);
    score[i] = s * 0.14142135623730951f;   // 1/sqrt(50)
}

__global__ void k_segmax(const int* __restrict__ ei, const float* __restrict__ score,
                         unsigned* __restrict__ mmax)
{
    size_t i = (size_t)blockIdx.x * blockDim.x + threadIdx.x;
    if (i >= (size_t)ETOT * NHEADS) return;
    int e = (int)(i >> 2), h = (int)(i & 3);
    int src = edge_src(ei, e);
    atomicMax(&mmax[(size_t)src * NHEADS + h], enc_f32(score[i]));
}

__global__ void k_expsum(const int* __restrict__ ei, const float* __restrict__ score,
                         const unsigned* __restrict__ mmax,
                         float* __restrict__ eexp, float* __restrict__ ssum,
                         float* __restrict__ cnt)
{
    size_t i = (size_t)blockIdx.x * blockDim.x + threadIdx.x;
    if (i >= (size_t)ETOT * NHEADS) return;
    int e = (int)(i >> 2), h = (int)(i & 3);
    int src = edge_src(ei, e);
    float m  = dec_f32(mmax[(size_t)src * NHEADS + h]);
    if (!isfinite(m)) m = 0.0f;
    float ex = __expf(score[i] - m);
    eexp[i] = ex;
    atomicAdd(&ssum[(size_t)src * NHEADS + h], ex);
    if (h == 0) atomicAdd(&cnt[src], 1.0f);
}

__global__ void k_alpha(const int* __restrict__ ei, const float* __restrict__ eexp,
                        const float* __restrict__ ssum, const float* __restrict__ cnt,
                        float* __restrict__ alpha)
{
    size_t i = (size_t)blockIdx.x * blockDim.x + threadIdx.x;
    if (i >= (size_t)ETOT * NHEADS) return;
    int e = (int)(i >> 2), h = (int)(i & 3);
    int src = edge_src(ei, e);
    alpha[i] = eexp[i] / (ssum[(size_t)src * NHEADS + h] + 1e-16f) * cnt[src];
}

// aggr[dst, d] += (msgn[src,d] + msge[e,d]) * alpha[e, d/DPH]
__global__ void k_aggregate(const int* __restrict__ ei, const float* __restrict__ alpha,
                            const float* __restrict__ msgn, const bf16_t* __restrict__ msge,
                            float* __restrict__ aggr)
{
    size_t i = (size_t)blockIdx.x * blockDim.x + threadIdx.x;
    if (i >= (size_t)ETOT * DD) return;
    int e = (int)(i / DD), d = (int)(i % DD);
    int src = edge_src(ei, e), dst = edge_dst(ei, e);
    float a = alpha[(size_t)e * NHEADS + d / DPH];
    float m = msgn[(size_t)src * DD + d] + (float)msge[i];
    atomicAdd(&aggr[(size_t)dst * DD + d], m * a);
}

// sens = max_n ||out[n,:]|| (non-negative -> raw-bit atomicMax is monotone)
__global__ void k_sens(const float* __restrict__ out, unsigned* __restrict__ sens)
{
    int n = blockIdx.x * blockDim.x + threadIdx.x;
    if (n >= NN) return;
    float s = 0.0f;
    const float* r = out + (size_t)n * DD;
    for (int d = 0; d < DD; ++d) s += r[d] * r[d];
    atomicMax(sens, __float_as_uint(sqrtf(s)));
}

__global__ void k_gelu_bf16(const float* __restrict__ in, bf16_t* __restrict__ out, size_t n)
{
    size_t i = (size_t)blockIdx.x * blockDim.x + threadIdx.x;
    if (i >= n) return;
    float x = in[i];
    out[i] = (bf16_t)(0.5f * x * (1.0f + erff(x * 0.70710678118654752f)));
}

// per-node: normalize Q row, dot with batch context key, add Laplace noise
__global__ void k_att(const float* __restrict__ Q, const float* __restrict__ Katt,
                      const unsigned* __restrict__ sens, float* __restrict__ att, int li)
{
    int n = blockIdx.x * blockDim.x + threadIdx.x;
    if (n >= NN) return;
    int b = n / N_NODE;
    const float* q = Q    + (size_t)n * DD;
    const float* k = Katt + (size_t)b * DD;
    float sq = 0.0f, dot = 0.0f;
    for (int d = 0; d < DD; ++d) { sq += q[d] * q[d]; dot += q[d] * k[d]; }
    float nrm = fmaxf(sqrtf(sq), 1e-12f);
    float a = (dot / nrm) * 0.07071067811865475f;   // 1/sqrt(200)
    float scale = __uint_as_float(*sens) * 4.0f;     // sens / (EP1/4)
    unsigned hsh = hash_u32(0x9e3779b9u ^ (unsigned)(li * 1664525) ^ ((unsigned)n * 2654435761u));
    float u  = (float)(hsh >> 8) * (1.0f / 16777216.0f) - 0.5f;
    float lap = -copysignf(__logf(fmaxf(1.0f - 2.0f * fabsf(u), 1e-12f)), u);
    att[n] = a + scale * lap;
}

// X[n,d] = att[n] * V[batch(n), d]
__global__ void k_xnext(const float* __restrict__ att, const float* __restrict__ Vatt,
                        float* __restrict__ Xcur)
{
    size_t i = (size_t)blockIdx.x * blockDim.x + threadIdx.x;
    if (i >= (size_t)NN * DD) return;
    int n = (int)(i / DD), d = (int)(i % DD);
    Xcur[i] = att[n] * Vatt[(size_t)(n / N_NODE) * DD + d];
}

// tiny dense C[M,N] = A@W + b, row stride of A = strideA (fp32, M small)
__global__ void k_dense_small(const float* __restrict__ A, int strideA,
                              const float* __restrict__ W, const float* __restrict__ b,
                              float* __restrict__ C, int M, int N, int K)
{
    int i = blockIdx.x * blockDim.x + threadIdx.x;
    if (i >= M * N) return;
    int m = i / N, n = i % N;
    float s = b ? b[n] : 0.0f;
    for (int k = 0; k < K; ++k) s += A[(size_t)m * strideA + k] * W[(size_t)k * N + n];
    C[i] = s;
}

// ---------------- host-side orchestration ----------------
static inline void ew(hipStream_t s, size_t n, void (*f)(const float*, bf16_t*, size_t),
                      const float* a, bf16_t* b) {
    f<<<(unsigned)((n + 255) / 256), 256, 0, s>>>(a, b, n);
}

static inline void launch_gemm(hipStream_t s, const bf16_t* A, int lda,
                               const bf16_t* Bt, const float* bias,
                               float* Cf, bf16_t* Cbf, int ldc,
                               int M, int N, int K, int act) {
    dim3 g(M / BM, (N + BN - 1) / BN);
    if (K == 200)
        gemm_bf16_wmma<200><<<g, 256, 0, s>>>(A, lda, Bt, bias, Cf, Cbf, ldc, N, act);
    else // K == 400
        gemm_bf16_wmma<400><<<g, 256, 0, s>>>(A, lda, Bt, bias, Cf, Cbf, ldc, N, act);
}

extern "C" void kernel_launch(void* const* d_in, const int* in_sizes, int n_in,
                              void* d_out, int out_size, void* d_ws, size_t ws_size,
                              hipStream_t stream)
{
    // inputs (setup_inputs order)
    const float* hidden = (const float*)d_in[0];
    const float* Xin    = (const float*)d_in[1];
    const float* nfe    = (const float*)d_in[2];
    const int*   ei     = (const int*)d_in[3];
    const int*   etype  = (const int*)d_in[4];
    const int*   ntype  = (const int*)d_in[5];
    const float* We1  = (const float*)d_in[7];  const float* be1  = (const float*)d_in[8];
    const float* We2  = (const float*)d_in[9];  const float* be2  = (const float*)d_in[10];
    const float* Wkey = (const float*)d_in[11]; const float* bkey = (const float*)d_in[12];
    const float* Wmsg = (const float*)d_in[13]; const float* bmsg = (const float*)d_in[14];
    const float* Wqry = (const float*)d_in[15]; const float* bqry = (const float*)d_in[16];
    const float* Wm1  = (const float*)d_in[17]; const float* bm1  = (const float*)d_in[18];
    const float* Wm2  = (const float*)d_in[19]; const float* bm2  = (const float*)d_in[20];
    const float* Wqa  = (const float*)d_in[21]; const float* bqa  = (const float*)d_in[22];
    const float* Wka  = (const float*)d_in[23]; const float* bka  = (const float*)d_in[24];
    const float* Wva  = (const float*)d_in[25]; const float* bva  = (const float*)d_in[26];
    const float* Wpr  = (const float*)d_in[27]; const float* bpr  = (const float*)d_in[28];

    // bump allocator on d_ws
    char* base = (char*)d_ws;
    size_t off = 0;
    auto alloc = [&](size_t bytes) -> void* {
        void* p = base + off;
        off = (off + bytes + 255) & ~(size_t)255;
        return p;
    };

    float*   Xcur    = (float*)  alloc((size_t)NN * DD * 4);
    bf16_t*  H1bf    = (bf16_t*) alloc((size_t)ETOT * DD * 2);   // reused as keye_bf
    bf16_t*  keye_bf = H1bf;                                      // H1 dead after edge GEMM
    bf16_t*  emb_bf  = (bf16_t*) alloc((size_t)ETOT * DD * 2);
    bf16_t*  msge_bf = (bf16_t*) alloc((size_t)ETOT * DD * 2);
    bf16_t*  xcat_bf = (bf16_t*) alloc((size_t)NN * 2 * DD * 2);
    float*   keyn    = (float*)  alloc((size_t)NN * DD * 4);
    float*   msgn    = (float*)  alloc((size_t)NN * DD * 4);
    float*   qryn    = (float*)  alloc((size_t)NN * DD * 4);
    float*   score   = (float*)  alloc((size_t)ETOT * NHEADS * 4);
    float*   eexp    = (float*)  alloc((size_t)ETOT * NHEADS * 4);
    float*   alpha   = (float*)  alloc((size_t)ETOT * NHEADS * 4);
    unsigned* mmax   = (unsigned*)alloc((size_t)NN * NHEADS * 4);
    float*   ssum    = (float*)  alloc((size_t)NN * NHEADS * 4);
    float*   cnt     = (float*)  alloc((size_t)NN * 4);
    float*   aggr    = (float*)  alloc((size_t)NN * DD * 4);
    bf16_t*  aggr_bf = (bf16_t*) alloc((size_t)NN * DD * 2);
    bf16_t*  hid_bf  = (bf16_t*) alloc((size_t)NN * DD * 2);
    float*   outb    = (float*)  alloc((size_t)NN * DD * 4);
    bf16_t*  xg_bf   = (bf16_t*) alloc((size_t)NN * DD * 2);
    float*   Qf      = (float*)  alloc((size_t)NN * DD * 4);
    float*   attb    = (float*)  alloc((size_t)NN * 4);
    unsigned* sens   = (unsigned*)alloc(256);
    float*   ctx     = (float*)  alloc((size_t)NBS * DD * 4);
    float*   Katt    = (float*)  alloc((size_t)NBS * DD * 4);
    float*   Vatt    = (float*)  alloc((size_t)NBS * DD * 4);
    // transposed + padded weights (Bt layout [NP][KP])
    bf16_t*  We2t    = (bf16_t*) alloc((size_t)NP * KP200 * 2);
    bf16_t*  Wkeynt  = (bf16_t*) alloc((size_t)KLAYERS * NP * KP400 * 2);
    bf16_t*  Wkeyet  = (bf16_t*) alloc((size_t)KLAYERS * NP * KP200 * 2);
    bf16_t*  Wmsgnt  = (bf16_t*) alloc((size_t)KLAYERS * NP * KP400 * 2);
    bf16_t*  Wmsget  = (bf16_t*) alloc((size_t)KLAYERS * NP * KP200 * 2);
    bf16_t*  Wqrynt  = (bf16_t*) alloc((size_t)KLAYERS * NP * KP400 * 2);
    bf16_t*  Wm1t    = (bf16_t*) alloc((size_t)KLAYERS * NP * KP200 * 2);
    bf16_t*  Wm2t    = (bf16_t*) alloc((size_t)KLAYERS * NP * KP200 * 2);
    bf16_t*  Wqat    = (bf16_t*) alloc((size_t)NP * KP200 * 2);
    (void)ws_size; (void)n_in; (void)in_sizes; (void)out_size;

    // 1) weight transpose+pad (cheap, every call => stateless)
    auto tpose = [&](const float* W, bf16_t* Bt, int K) {
        int KPAD = (K == 200) ? KP200 : KP400;
        size_t n = (size_t)NP * KPAD;
        k_transpose_pad<<<(unsigned)((n + 255) / 256), 256, 0, stream>>>(W, Bt, K, DD, KPAD, n);
    };
    tpose(We2, We2t, 200);
    tpose(Wqa, Wqat, 200);
    for (int li = 0; li < KLAYERS; ++li) {
        tpose(Wkey + (size_t)li * 600 * DD,         Wkeynt + (size_t)li * NP * KP400, 400);
        tpose(Wkey + ((size_t)li * 600 + 400) * DD, Wkeyet + (size_t)li * NP * KP200, 200);
        tpose(Wmsg + (size_t)li * 600 * DD,         Wmsgnt + (size_t)li * NP * KP400, 400);
        tpose(Wmsg + ((size_t)li * 600 + 400) * DD, Wmsget + (size_t)li * NP * KP200, 200);
        tpose(Wqry + (size_t)li * 400 * DD,         Wqrynt + (size_t)li * NP * KP400, 400);
        tpose(Wm1  + (size_t)li * DD * DD,          Wm1t   + (size_t)li * NP * KP200, 200);
        tpose(Wm2  + (size_t)li * DD * DD,          Wm2t   + (size_t)li * NP * KP200, 200);
    }

    // 2) edge encoder: H1 = relu(onehot @ We1 + be1) via gather; emb = H1 @ We2 + be2
    {
        size_t n = (size_t)ETOT * DD;
        k_edge_encode<<<(unsigned)((n + 255) / 256), 256, 0, stream>>>(ei, etype, ntype, We1, be1, H1bf);
        launch_gemm(stream, H1bf, DD, We2t, be2, nullptr, emb_bf, DD, ETOT, DD, 200, 0);
    }

    // 3) context path (tiny): ctx = hs[:,0,:] @ Wproj + b; K/V = ctx @ Wk/v + b
    k_dense_small<<<(NBS * DD + 255) / 256, 256, 0, stream>>>(hidden, SEQL * HSZ, Wpr, bpr, ctx,  NBS, DD, HSZ);
    k_dense_small<<<(NBS * DD + 255) / 256, 256, 0, stream>>>(ctx,    DD,         Wka, bka, Katt, NBS, DD, DD);
    k_dense_small<<<(NBS * DD + 255) / 256, 256, 0, stream>>>(ctx,    DD,         Wva, bva, Vatt, NBS, DD, DD);

    // 4) X working copy
    k_copy_f32<<<(unsigned)(((size_t)NN * DD + 255) / 256), 256, 0, stream>>>(Xin, Xcur, (size_t)NN * DD);

    const size_t nEH = (size_t)ETOT * NHEADS;
    const size_t nED = (size_t)ETOT * DD;
    const size_t nND = (size_t)NN * DD;

    for (int li = 0; li < KLAYERS; ++li) {
        // a) xcat (bf16) from current X + node_feature_extra
        k_xcat<<<(unsigned)(((size_t)NN * 2 * DD + 255) / 256), 256, 0, stream>>>(Xcur, nfe, xcat_bf);

        // b) node-side GEMMs (bias folded here), edge-side GEMMs (no bias)
        launch_gemm(stream, xcat_bf, 2 * DD, Wkeynt + (size_t)li * NP * KP400, bkey + li * DD, keyn, nullptr, DD, NN,   DD, 400, 0);
        launch_gemm(stream, xcat_bf, 2 * DD, Wmsgnt + (size_t)li * NP * KP400, bmsg + li * DD, msgn, nullptr, DD, NN,   DD, 400, 0);
        launch_gemm(stream, xcat_bf, 2 * DD, Wqrynt + (size_t)li * NP * KP400, bqry + li * DD, qryn, nullptr, DD, NN,   DD, 400, 0);
        launch_gemm(stream, emb_bf,  DD,     Wkeyet + (size_t)li * NP * KP200, nullptr, nullptr, keye_bf, DD, ETOT, DD, 200, 0);
        launch_gemm(stream, emb_bf,  DD,     Wmsget + (size_t)li * NP * KP200, nullptr, nullptr, msge_bf, DD, ETOT, DD, 200, 0);

        // c) segment softmax over src + degree scaling
        k_fill_u32<<<(unsigned)(((size_t)NN * NHEADS + 255) / 256), 256, 0, stream>>>(mmax, 0x007FFFFFu, (size_t)NN * NHEADS); // enc(-inf)
        k_fill_f32<<<(unsigned)(((size_t)NN * NHEADS + 255) / 256), 256, 0, stream>>>(ssum, 0.0f, (size_t)NN * NHEADS);
        k_fill_f32<<<(NN + 255) / 256, 256, 0, stream>>>(cnt, 0.0f, (size_t)NN);
        k_fill_f32<<<(unsigned)((nND + 255) / 256), 256, 0, stream>>>(aggr, 0.0f, nND);
        k_score    <<<(unsigned)((nEH + 255) / 256), 256, 0, stream>>>(ei, qryn, keyn, keye_bf, score);
        k_segmax   <<<(unsigned)((nEH + 255) / 256), 256, 0, stream>>>(ei, score, mmax);
        k_expsum   <<<(unsigned)((nEH + 255) / 256), 256, 0, stream>>>(ei, score, mmax, eexp, ssum, cnt);
        k_alpha    <<<(unsigned)((nEH + 255) / 256), 256, 0, stream>>>(ei, eexp, ssum, cnt, alpha);
        k_aggregate<<<(unsigned)((nED + 255) / 256), 256, 0, stream>>>(ei, alpha, msgn, msge_bf, aggr);

        // d) MLP + gelu
        ew(stream, nND, k_f32_to_bf16, aggr, aggr_bf);
        launch_gemm(stream, aggr_bf, DD, Wm1t + (size_t)li * NP * KP200, bm1 + li * DD,
                    nullptr, hid_bf, DD, NN, DD, 200, 1);
        launch_gemm(stream, hid_bf,  DD, Wm2t + (size_t)li * NP * KP200, bm2 + li * DD,
                    outb, nullptr, DD, NN, DD, 200, 0);
        k_fill_u32<<<1, 1, 0, stream>>>(sens, 0u, 1);
        k_sens<<<(NN + 255) / 256, 256, 0, stream>>>(outb, sens);
        k_gelu_bf16<<<(unsigned)((nND + 255) / 256), 256, 0, stream>>>(outb, xg_bf, nND);

        // e) context attention + DP noise; X <- att * V
        launch_gemm(stream, xg_bf, DD, Wqat, bqa, Qf, nullptr, DD, NN, DD, 200, 0);
        k_att  <<<(NN + 255) / 256, 256, 0, stream>>>(Qf, Katt, sens, attb, li);
        k_xnext<<<(unsigned)((nND + 255) / 256), 256, 0, stream>>>(attb, Vatt, Xcur);
    }

    // 5) output
    k_copy_f32<<<(unsigned)((nND + 255) / 256), 256, 0, stream>>>(Xcur, (float*)d_out, nND);
}
